// GraMDTA_82171314307433
// MI455X (gfx1250) — matrix operand order repaired and checked by reading, not compiled, not run on hardware
//
#include <hip/hip_runtime.h>
#include <math.h>

// ---------------- model constants ----------------
#define NNODES 25600
#define NEDGES 102400
#define NAUG   128000      // edges + self loops
#define BB     512
#define NPG    50
#define FD     78
#define H1     10
#define HD1    780         // H1*FD
#define OUT    256
#define EDGED  10
#define SEQL   1000
#define MAXC   26
#define CNNDIM 7392

// input indices (setup_inputs dict order, params flattened in insertion order)
enum {
  IN_X = 0, IN_EA = 1,
  P_G1_W, P_G1_AS, P_G1_AD, P_G1_WE, P_G1_AE, P_G1_B,
  P_G2_W, P_G2_AS, P_G2_AD, P_G2_WE, P_G2_AE, P_G2_B,
  P_MOL_W, P_MOL_B, P_EMB_XT,
  P_C1W, P_C1B, P_C2W, P_C2B, P_C3W, P_C3B, P_C4W, P_C4B, P_C5W, P_C5B, P_C6W, P_C6B,
  P_PROT_W, P_PROT_B, P_D_EMB, P_P_EMB, P_D_W, P_D_B, P_P_W, P_P_B,
  P_AIN_W, P_AIN_B, P_AOUT_W, P_AOUT_B,
  P_N1_G, P_N1_B,
  P_FC1_W, P_FC1_B, P_FC2_W, P_FC2_B, P_FC3_W, P_FC3_B, P_OUT_W, P_OUT_B,
  IN_ESRC, IN_EDST, IN_BATCH, IN_DKG, IN_PKG, IN_SEQ
};

typedef __attribute__((ext_vector_type(16))) _Float16 v16h;
typedef __attribute__((ext_vector_type(8)))  _Float16 v8h;
typedef __attribute__((ext_vector_type(2)))  _Float16 v2h;
typedef __attribute__((ext_vector_type(8)))  float    v8f;

// ---------------- small device helpers ----------------
__device__ __forceinline__ void atomicMaxF(float* addr, float v) {
  if (v >= 0.f) atomicMax((int*)addr, __float_as_int(v));
  else          atomicMin((unsigned int*)addr, __float_as_uint(v));
}

__device__ __forceinline__ v16h ld_frag(const _Float16* p0, const _Float16* p1) {
  v8h lo = *(const v8h*)p0;   // 16B aligned -> ds_load_b128
  v8h hi = *(const v8h*)p1;
  return __builtin_shufflevector(lo, hi, 0,1,2,3,4,5,6,7,8,9,10,11,12,13,14,15);
}

#define BM 64
#define BN 64
#define BKK 32
#define APAD 8   // halves; row = 80B keeps every fragment offset 16B aligned

// ---------------- LDS-tiled WMMA GEMM ----------------
// C[M,N] = act( A[M,K] * B + bias[n] ), B row-major [K,N] (transB=0) or [N,K] (transB=1)
// block = 256 threads = 8 waves; tile 64x64x32; each wave: 16 rows x 32 cols (2 wmma)
__global__ __launch_bounds__(256) void gemm_wmma_tiled(
    const float* __restrict__ A, int lda,
    const float* __restrict__ B, int ldb, int transB,
    float* __restrict__ C, int ldc,
    int M, int N, int K,
    const float* __restrict__ bias, int act)
{
  __shared__ __align__(16) _Float16 As[BM][BKK + APAD];
  __shared__ __align__(16) _Float16 Bs[BN][BKK + APAD];   // transposed: [n][k]
  int tid  = threadIdx.x;
  int bm   = blockIdx.y * BM;
  int bn   = blockIdx.x * BN;
  int wave = tid >> 5, lane = tid & 31;
  int wm = wave >> 1, wn = wave & 1;        // 4x2 wave grid
  int r = lane & 15, hi = lane >> 4;
  bool interior = (bm + BM <= M) && (bn + BN <= N);

  v8f acc0 = {}, acc1 = {};

  for (int k0 = 0; k0 < K; k0 += BKK) {
    if (interior && (k0 + BKK <= K)) {
      // ---------- fast path: unguarded two-phase staging ----------
      {
        const float* Ap = A + (long)bm * lda + k0;
        int c2 = (tid & 15) * 2;
        float2 av[4];
#pragma unroll
        for (int j = 0; j < 4; ++j) {               // phase 1: issue all loads
          int row = (tid >> 4) + j * 16;            // 0..63
          av[j].x = Ap[(long)row * lda + c2];
          av[j].y = Ap[(long)row * lda + c2 + 1];
        }
#pragma unroll
        for (int j = 0; j < 4; ++j) {               // phase 2: convert + store
          int row = (tid >> 4) + j * 16;
          v2h p; p[0] = (_Float16)av[j].x; p[1] = (_Float16)av[j].y;
          *(v2h*)&As[row][c2] = p;                  // ds_store_b32
        }
      }
      if (!transB) {
        const float* Bp = B + (long)k0 * ldb + bn;
        int n = tid & 63;
        float bv[8];
#pragma unroll
        for (int j = 0; j < 8; ++j) {
          int kk = (tid >> 6) + j * 4;              // 0..31
          bv[j] = Bp[(long)kk * ldb + n];
        }
#pragma unroll
        for (int j = 0; j < 8; ++j) {
          int kk = (tid >> 6) + j * 4;
          Bs[n][kk] = (_Float16)bv[j];
        }
      } else {
        const float* Bp = B + (long)bn * ldb + k0;
        int c2 = (tid & 15) * 2;
        float2 bv[4];
#pragma unroll
        for (int j = 0; j < 4; ++j) {
          int n = (tid >> 4) + j * 16;              // 0..63
          bv[j].x = Bp[(long)n * ldb + c2];
          bv[j].y = Bp[(long)n * ldb + c2 + 1];
        }
#pragma unroll
        for (int j = 0; j < 4; ++j) {
          int n = (tid >> 4) + j * 16;
          v2h p; p[0] = (_Float16)bv[j].x; p[1] = (_Float16)bv[j].y;
          *(v2h*)&Bs[n][c2] = p;
        }
      }
    } else {
      // ---------- guarded path (edge tiles / K tail) ----------
#pragma unroll
      for (int j = 0; j < 8; ++j) {
        int row = (tid >> 5) + j * 8;
        int col = tid & 31;
        int gm = bm + row, gk = k0 + col;
        float v = A[(long)min(gm, M - 1) * lda + min(gk, K - 1)];
        v = (gm < M && gk < K) ? v : 0.f;
        As[row][col] = (_Float16)v;
      }
      if (!transB) {
#pragma unroll
        for (int j = 0; j < 8; ++j) {
          int n  = tid & 63;
          int kk = (tid >> 6) + j * 4;
          int gk = k0 + kk, gn = bn + n;
          float v = B[(long)min(gk, K - 1) * ldb + min(gn, N - 1)];
          v = (gk < K && gn < N) ? v : 0.f;
          Bs[n][kk] = (_Float16)v;
        }
      } else {
#pragma unroll
        for (int j = 0; j < 8; ++j) {
          int kk = tid & 31;
          int n  = (tid >> 5) + j * 8;
          int gk = k0 + kk, gn = bn + n;
          float v = B[(long)min(gn, N - 1) * ldb + min(gk, K - 1)];
          v = (gk < K && gn < N) ? v : 0.f;
          Bs[n][kk] = (_Float16)v;
        }
      }
    }
    __syncthreads();
    // ---- fragments from LDS (ds_load_b128) ----
    const _Float16* ar = &As[wm * 16 + r][0];
    v16h af  = ld_frag(ar + hi * 8, ar + 16 + hi * 8);
    const _Float16* b0 = &Bs[wn * 32 + r][hi * 8];
    const _Float16* b1 = &Bs[wn * 32 + 16 + r][hi * 8];
    v16h bf0 = ld_frag(b0, b0 + 16);
    v16h bf1 = ld_frag(b1, b1 + 16);
    acc0 = __builtin_amdgcn_wmma_f32_16x16x32_f16(false, af, false, bf0, (short)0, acc0, false, false);
    acc1 = __builtin_amdgcn_wmma_f32_16x16x32_f16(false, af, false, bf1, (short)0, acc1, false, false);
    __syncthreads();
  }
  // ---- store ----
  int cn0 = bn + wn * 32 + (lane & 15);
  int cn1 = cn0 + 16;
  if (interior) {
#pragma unroll
    for (int rr = 0; rr < 8; ++rr) {
      int cm = bm + wm * 16 + hi * 8 + rr;
      float v0 = acc0[rr], v1 = acc1[rr];
      if (bias) { v0 += bias[cn0]; v1 += bias[cn1]; }
      if (act == 1)      { v0 = fmaxf(v0, 0.f); v1 = fmaxf(v1, 0.f); }
      else if (act == 2) { v0 = v0 > 0.f ? v0 : expm1f(v0); v1 = v1 > 0.f ? v1 : expm1f(v1); }
      C[(long)cm * ldc + cn0] = v0;
      C[(long)cm * ldc + cn1] = v1;
    }
  } else {
#pragma unroll
    for (int rr = 0; rr < 8; ++rr) {
      int cm = bm + wm * 16 + hi * 8 + rr;
      if (cm < M) {
        if (cn0 < N) {
          float v = acc0[rr];
          if (bias) v += bias[cn0];
          if (act == 1)      v = fmaxf(v, 0.f);
          else if (act == 2) v = v > 0.f ? v : expm1f(v);
          C[(long)cm * ldc + cn0] = v;
        }
        if (cn1 < N) {
          float v = acc1[rr];
          if (bias) v += bias[cn1];
          if (act == 1)      v = fmaxf(v, 0.f);
          else if (act == 2) v = v > 0.f ? v : expm1f(v);
          C[(long)cm * ldc + cn1] = v;
        }
      }
    }
  }
}

// ---------------- implicit-GEMM WMMA conv1d ----------------
// y[b,o,t] = act( sum_{i,kk} w[o,i,kk] * x[b,i,t+kk] + bias[o] )
// A = w [Cout][Cin*Kk] (row-major already), B im2col-gathered while staging.
// grid: x = ceil(Lout/64), y = ceil(Cout/64), z = batch. K=Cin*Kk is always a
// multiple of 32 here, so only M/N fringes use the guarded path.
__global__ __launch_bounds__(256) void conv1d_wmma(
    const float* __restrict__ x, const float* __restrict__ w,
    const float* __restrict__ bias, float* __restrict__ y,
    int Cin, int Lin, int Cout, int Kk, int Lout, int act)
{
  __shared__ __align__(16) _Float16 As[BM][BKK + APAD];
  __shared__ __align__(16) _Float16 Bs[BN][BKK + APAD];   // [t][k]
  int tid = threadIdx.x;
  int b   = blockIdx.z;
  int bm  = blockIdx.y * BM;      // Cout
  int bn  = blockIdx.x * BN;      // Lout
  int K   = Cin * Kk;
  const float* xb = x + (long)b * Cin * Lin;
  int wave = tid >> 5, lane = tid & 31;
  int wm = wave >> 1, wn = wave & 1;
  int r = lane & 15, hi = lane >> 4;
  bool interior = (bm + BM <= Cout) && (bn + BN <= Lout);

  v8f acc0 = {}, acc1 = {};

  for (int k0 = 0; k0 < K; k0 += BKK) {
    if (interior && (k0 + BKK <= K)) {
      {
        const float* Ap = w + (long)bm * K + k0;
        int c2 = (tid & 15) * 2;
        float2 av[4];
#pragma unroll
        for (int j = 0; j < 4; ++j) {
          int row = (tid >> 4) + j * 16;
          av[j].x = Ap[(long)row * K + c2];
          av[j].y = Ap[(long)row * K + c2 + 1];
        }
#pragma unroll
        for (int j = 0; j < 4; ++j) {
          int row = (tid >> 4) + j * 16;
          v2h p; p[0] = (_Float16)av[j].x; p[1] = (_Float16)av[j].y;
          *(v2h*)&As[row][c2] = p;
        }
      }
      {
        int n = tid & 63;
        float bv[8];
#pragma unroll
        for (int j = 0; j < 8; ++j) {
          int kg = k0 + (tid >> 6) + j * 4;
          int ic = kg / Kk, kr = kg - ic * Kk;
          bv[j] = xb[(long)ic * Lin + (bn + n) + kr];
        }
#pragma unroll
        for (int j = 0; j < 8; ++j) {
          int kk = (tid >> 6) + j * 4;
          Bs[n][kk] = (_Float16)bv[j];
        }
      }
    } else {
#pragma unroll
      for (int j = 0; j < 8; ++j) {
        int row = (tid >> 5) + j * 8;
        int col = tid & 31;
        int gm = bm + row, gk = k0 + col;
        float v = w[(long)min(gm, Cout - 1) * K + min(gk, K - 1)];
        v = (gm < Cout && gk < K) ? v : 0.f;
        As[row][col] = (_Float16)v;
      }
#pragma unroll
      for (int j = 0; j < 8; ++j) {
        int n  = tid & 63;
        int kk = (tid >> 6) + j * 4;
        int kg = min(k0 + kk, K - 1);
        int ic = kg / Kk, kr = kg - ic * Kk;
        int t  = min(bn + n, Lout - 1);
        float v = xb[(long)ic * Lin + t + kr];
        v = ((k0 + kk) < K && (bn + n) < Lout) ? v : 0.f;
        Bs[n][kk] = (_Float16)v;
      }
    }
    __syncthreads();
    const _Float16* ar = &As[wm * 16 + r][0];
    v16h af  = ld_frag(ar + hi * 8, ar + 16 + hi * 8);
    const _Float16* b0 = &Bs[wn * 32 + r][hi * 8];
    const _Float16* b1 = &Bs[wn * 32 + 16 + r][hi * 8];
    v16h bf0 = ld_frag(b0, b0 + 16);
    v16h bf1 = ld_frag(b1, b1 + 16);
    acc0 = __builtin_amdgcn_wmma_f32_16x16x32_f16(false, af, false, bf0, (short)0, acc0, false, false);
    acc1 = __builtin_amdgcn_wmma_f32_16x16x32_f16(false, af, false, bf1, (short)0, acc1, false, false);
    __syncthreads();
  }
  int t0 = bn + wn * 32 + (lane & 15);
  int t1 = t0 + 16;
  if (interior) {
#pragma unroll
    for (int rr = 0; rr < 8; ++rr) {
      int o = bm + wm * 16 + hi * 8 + rr;
      float bo = bias[o];
      float v0 = acc0[rr] + bo, v1 = acc1[rr] + bo;
      if (act) { v0 = fmaxf(v0, 0.f); v1 = fmaxf(v1, 0.f); }
      y[((long)b * Cout + o) * Lout + t0] = v0;
      y[((long)b * Cout + o) * Lout + t1] = v1;
    }
  } else {
#pragma unroll
    for (int rr = 0; rr < 8; ++rr) {
      int o = bm + wm * 16 + hi * 8 + rr;
      if (o < Cout) {
        float bo = bias[o];
        if (t0 < Lout) {
          float v = acc0[rr] + bo;
          if (act) v = fmaxf(v, 0.f);
          y[((long)b * Cout + o) * Lout + t0] = v;
        }
        if (t1 < Lout) {
          float v = acc1[rr] + bo;
          if (act) v = fmaxf(v, 0.f);
          y[((long)b * Cout + o) * Lout + t1] = v;
        }
      }
    }
  }
}

// ---------------- misc kernels ----------------
__global__ void fill_f32(float* p, float v, long n) {
  long i = (long)blockIdx.x * blockDim.x + threadIdx.x;
  if (i < n) p[i] = v;
}

// out[k*H+h] = sum_d W[(k*H+h)*D + d] * a[h*D + d]
__global__ void fold_aw(const float* __restrict__ W, const float* __restrict__ a,
                        float* __restrict__ out, int K, int H, int D) {
  int i = blockIdx.x * blockDim.x + threadIdx.x;
  if (i >= K * H) return;
  int h = i % H, k = i / H;
  float s = 0.f;
  for (int d = 0; d < D; ++d) s += W[((long)k * H + h) * D + d] * a[h * D + d];
  out[i] = s;
}

__global__ void edge_accum(const float* __restrict__ ea, const int* __restrict__ dst,
                           float* __restrict__ s, float* __restrict__ cnt) {
  int i = blockIdx.x * blockDim.x + threadIdx.x;
  if (i >= NEDGES * EDGED) return;
  int c = i % EDGED, e = i / EDGED;
  int d = dst[e];
  atomicAdd(&s[(long)d * EDGED + c], ea[i]);
  if (c == 0) atomicAdd(&cnt[d], 1.f);
}

__global__ void build_ea_aug(const float* __restrict__ ea, const float* __restrict__ s,
                             const float* __restrict__ cnt, float* __restrict__ out) {
  int i = blockIdx.x * blockDim.x + threadIdx.x;
  if (i >= NAUG * EDGED) return;
  int c = i % EDGED, e = i / EDGED;
  if (e < NEDGES) out[i] = ea[i];
  else {
    int n = e - NEDGES;
    out[i] = s[(long)n * EDGED + c] / fmaxf(cnt[n], 1.f);
  }
}

__global__ void gat_alpha(const float* __restrict__ asrc, const float* __restrict__ adst,
                          const float* __restrict__ ae, const int* __restrict__ esrc,
                          const int* __restrict__ edst, int E, int Eaug, int H,
                          float* __restrict__ alpha, float* __restrict__ mbuf) {
  long i = (long)blockIdx.x * blockDim.x + threadIdx.x;
  if (i >= (long)Eaug * H) return;
  int h = (int)(i % H); long e = i / H;
  int s = e < E ? esrc[e] : (int)(e - E);
  int d = e < E ? edst[e] : (int)(e - E);
  float v = asrc[(long)s * H + h] + adst[(long)d * H + h] + ae[i];
  v = v > 0.f ? v : 0.2f * v;         // leaky_relu 0.2
  alpha[i] = v;
  atomicMaxF(&mbuf[(long)d * H + h], v);
}

__global__ void gat_expa(float* __restrict__ alpha, const float* __restrict__ mbuf,
                         float* __restrict__ den, const int* __restrict__ edst,
                         int E, int Eaug, int H) {
  long i = (long)blockIdx.x * blockDim.x + threadIdx.x;
  if (i >= (long)Eaug * H) return;
  int h = (int)(i % H); long e = i / H;
  int d = e < E ? edst[e] : (int)(e - E);
  float v = expf(alpha[i] - mbuf[(long)d * H + h]);
  alpha[i] = v;
  atomicAdd(&den[(long)d * H + h], v);
}

// agg[dst, c] += hpre[src, c] * expa[e,h]/(den[dst,h]+eps), c = h*D+d
__global__ void gat_scatter(const float* __restrict__ hpre, const float* __restrict__ expa,
                            const float* __restrict__ den, const int* __restrict__ esrc,
                            const int* __restrict__ edst, int E, int Eaug, int H, int D,
                            float* __restrict__ agg) {
  long i = (long)blockIdx.x * blockDim.x + threadIdx.x;
  int HD = H * D;
  if (i >= (long)Eaug * HD) return;
  int c = (int)(i % HD); long e = i / HD;
  int h = c / D;
  int s = e < E ? esrc[e] : (int)(e - E);
  int d = e < E ? edst[e] : (int)(e - E);
  float w = expa[e * H + h] / (den[(long)d * H + h] + 1e-16f);
  atomicAdd(&agg[(long)d * HD + c], hpre[(long)s * HD + c] * w);
}

__global__ void bias_act(float* __restrict__ y, const float* __restrict__ bias,
                         long n, int D, int act) {
  long i = (long)blockIdx.x * blockDim.x + threadIdx.x;
  if (i >= n) return;
  float v = y[i] + bias[i % D];
  if (act == 1)      v = fmaxf(v, 0.f);
  else if (act == 2) v = v > 0.f ? v : expm1f(v);
  y[i] = v;
}

__global__ void pool_max(const float* __restrict__ h, float* __restrict__ xg) {
  int i = blockIdx.x * blockDim.x + threadIdx.x;
  if (i >= BB * OUT) return;
  int d = i & 255, b = i >> 8;
  const float* p = h + ((long)b * NPG) * OUT + d;
  float m = -1e30f;
  for (int k = 0; k < NPG; ++k) m = fmaxf(m, p[k * OUT]);
  xg[i] = m;
}

__global__ void gather_rows(const float* __restrict__ tab, const int* __restrict__ idx,
                            float* __restrict__ out, int R, int C) {
  int i = blockIdx.x * blockDim.x + threadIdx.x;
  if (i >= R * C) return;
  int c = i % C, r = i / C;
  out[i] = tab[(long)idx[r] * C + c];
}

// fused protein conv1: emb table (26x256) fully LDS-resident
#define EROW 272
__global__ __launch_bounds__(256) void conv1_fused(
    const int* __restrict__ seq, const float* __restrict__ emb,
    const float* __restrict__ w, const float* __restrict__ bias,
    float* __restrict__ y /* [512,32,249] */) {
  __shared__ float es[MAXC * EROW];
  __shared__ float wsh[8 * 32];     // [kk][o]
  int b = blockIdx.x, tid = threadIdx.x;
  for (int i = tid; i < MAXC * EROW; i += 256) {
    int row = i / EROW, col = i % EROW;
    es[i] = col < 256 ? emb[row * 256 + col] : 0.f;
  }
  __syncthreads();
  int o = tid & 31, tb = tid >> 5;
  float acc[32];
#pragma unroll
  for (int j = 0; j < 32; ++j) acc[j] = 0.f;
  for (int i = 0; i < SEQL; ++i) {
    __syncthreads();
    wsh[tb * 32 + o] = w[(long)o * 8000 + i * 8 + tb];  // c1_w[o,i,kk=tb]
    __syncthreads();
    int row = seq[b * SEQL + i];
    const float* er = &es[row * EROW];
#pragma unroll 4
    for (int j = 0; j < 32; ++j) {
      int t = tb + j * 8;
      float s = 0.f;
#pragma unroll
      for (int kk = 0; kk < 8; ++kk) s += wsh[kk * 32 + o] * er[t + kk];
      acc[j] += s;
    }
  }
  float bo = bias[o];
  for (int j = 0; j < 32; ++j) {
    int t = tb + j * 8;
    if (t < 249) y[((long)b * 32 + o) * 249 + t] = fmaxf(acc[j] + bo, 0.f);
  }
}

__global__ void build_xc(const float* __restrict__ a, const float* __restrict__ b,
                         const float* __restrict__ c, const float* __restrict__ d,
                         float* __restrict__ xc) {
  int i = blockIdx.x * blockDim.x + threadIdx.x;
  if (i >= BB * OUT) return;
  int col = i & 255, row = i >> 8;
  float* o = xc + (long)row * 1024;
  o[col] = a[i]; o[256 + col] = b[i]; o[512 + col] = c[i]; o[768 + col] = d[i];
}

// 4-token, 4-head attention; qkv [512,4,768] -> out [512,4,256]
__global__ void mha4(const float* __restrict__ qkv, float* __restrict__ out) {
  int i = blockIdx.x * blockDim.x + threadIdx.x;  // b*16 + hh*4 + qi
  if (i >= BB * 16) return;
  int qi = i & 3, hh = (i >> 2) & 3, b = i >> 4;
  const float* base = qkv + (long)b * 4 * 768;
  const float* q = base + qi * 768 + hh * 64;
  float sc[4]; float mx = -1e30f;
  for (int ki = 0; ki < 4; ++ki) {
    const float* k = base + ki * 768 + 256 + hh * 64;
    float s = 0.f;
    for (int d = 0; d < 64; ++d) s += q[d] * k[d];
    s *= 0.125f;                     // 1/sqrt(64)
    sc[ki] = s; mx = fmaxf(mx, s);
  }
  float den = 0.f;
  for (int ki = 0; ki < 4; ++ki) { sc[ki] = expf(sc[ki] - mx); den += sc[ki]; }
  float inv = 1.f / den;
  float* ob = out + ((long)b * 4 + qi) * 256 + hh * 64;
  for (int d = 0; d < 64; ++d) {
    float s = 0.f;
    for (int ki = 0; ki < 4; ++ki) s += sc[ki] * base[ki * 768 + 512 + hh * 64 + d];
    ob[d] = s * inv;
  }
}

// out = LN(xc + proj) * g + b, rows of 256
__global__ void add_ln(const float* __restrict__ xc, const float* __restrict__ proj,
                       const float* __restrict__ g, const float* __restrict__ bb,
                       float* __restrict__ out) {
  int row = blockIdx.x * blockDim.x + threadIdx.x;
  if (row >= BB * 4) return;
  const float* a = xc + (long)row * 256;
  const float* p = proj + (long)row * 256;
  float mu = 0.f;
  for (int i = 0; i < 256; ++i) mu += a[i] + p[i];
  mu *= (1.f / 256.f);
  float var = 0.f;
  for (int i = 0; i < 256; ++i) { float d = a[i] + p[i] - mu; var += d * d; }
  var *= (1.f / 256.f);
  float inv = rsqrtf(var + 1e-5f);
  float* o = out + (long)row * 256;
  for (int i = 0; i < 256; ++i) o[i] = (a[i] + p[i] - mu) * inv * g[i] + bb[i];
}

// ---------------- host side ----------------
static inline unsigned gb(long n) { return (unsigned)((n + 255) / 256); }

static void gemm(hipStream_t st, const float* A, int lda, const float* B, int ldb,
                 int tB, float* C, int ldc, int M, int N, int K,
                 const float* bias, int act) {
  dim3 g((unsigned)((N + BN - 1) / BN), (unsigned)((M + BM - 1) / BM), 1);
  gemm_wmma_tiled<<<g, 256, 0, st>>>(A, lda, B, ldb, tB, C, ldc, M, N, K, bias, act);
}

static void convw(hipStream_t st, const float* x, const float* w, const float* b,
                  float* y, int Cin, int Lin, int Cout, int Kk, int Lout, int act) {
  dim3 g((unsigned)((Lout + BN - 1) / BN), (unsigned)((Cout + BM - 1) / BM), BB);
  conv1d_wmma<<<g, 256, 0, st>>>(x, w, b, y, Cin, Lin, Cout, Kk, Lout, act);
}

extern "C" void kernel_launch(void* const* d_in, const int* in_sizes, int n_in,
                              void* d_out, int out_size, void* d_ws, size_t ws_size,
                              hipStream_t stream) {
  (void)in_sizes; (void)n_in; (void)out_size; (void)ws_size;
  const float* x   = (const float*)d_in[IN_X];
  const float* ea  = (const float*)d_in[IN_EA];
  const int* esrc  = (const int*)d_in[IN_ESRC];
  const int* edst  = (const int*)d_in[IN_EDST];
  const int* dkg   = (const int*)d_in[IN_DKG];
  const int* pkg   = (const int*)d_in[IN_PKG];
  const int* seq   = (const int*)d_in[IN_SEQ];
#define PF(i) ((const float*)d_in[i])
  float* outp = (float*)d_out;

  // workspace allocator
  char* wsb = (char*)d_ws; size_t off = 0;
  auto alloc = [&](long nfloats) -> float* {
    float* p = (float*)(wsb + off);
    off += (((size_t)nfloats * 4) + 255) & ~(size_t)255;
    return p;
  };
  float* ws1   = alloc(FD * H1);
  float* wd1   = alloc(FD * H1);
  float* we1   = alloc(EDGED * H1);
  float* wee2  = alloc(EDGED);
  float* sEA   = alloc((long)NNODES * EDGED);
  float* cEA   = alloc(NNODES);
  float* eaug  = alloc((long)NAUG * EDGED);
  float* A1    = alloc((long)NNODES * HD1);   // hpre1 ; later y3 / y5
  float* asrc1 = alloc((long)NNODES * H1);
  float* adst1 = alloc((long)NNODES * H1);
  float* ae1   = alloc((long)NAUG * H1);
  float* alp1  = alloc((long)NAUG * H1);
  float* m1    = alloc((long)NNODES * H1);
  float* den1  = alloc((long)NNODES * H1);
  float* B1    = alloc((long)NNODES * HD1);   // agg1/h1 ; later y4 / y6
  float* h2pre = alloc((long)NNODES * OUT);
  float* asrc2 = alloc(NNODES);
  float* adst2 = alloc(NNODES);
  float* ae2   = alloc(NAUG);
  float* alp2  = alloc(NAUG);
  float* m2    = alloc(NNODES);
  float* den2  = alloc(NNODES);
  float* agg2  = alloc((long)NNODES * OUT);   // h2 ; later y1
  float* xg    = alloc(BB * OUT);
  float* xmol  = alloc(BB * OUT);
  float* dgat  = alloc(BB * 128);
  float* pgat  = alloc(BB * 128);
  float* xdkg  = alloc(BB * OUT);
  float* xpkg  = alloc(BB * OUT);
  float* y2b   = alloc((long)BB * 64 * 245);
  float* xprot = alloc(BB * OUT);
  float* xc    = alloc(BB * 4 * OUT);
  float* qkv   = alloc((long)BB * 4 * 768);
  float* attno = alloc(BB * 4 * OUT);
  float* proj  = alloc(BB * 4 * OUT);
  float* xcln  = alloc(BB * 4 * OUT);         // == z [512,1024]
  float* z1    = alloc(BB * 1024);
  float* z2    = alloc(BB * 512);
  float* z3    = alloc(BB * 256);

  // ---- attention-weight folds ----
  fold_aw<<<gb(FD * H1), 256, 0, stream>>>(PF(P_G1_W),  PF(P_G1_AS), ws1, FD, H1, FD);
  fold_aw<<<gb(FD * H1), 256, 0, stream>>>(PF(P_G1_W),  PF(P_G1_AD), wd1, FD, H1, FD);
  fold_aw<<<gb(EDGED * H1), 256, 0, stream>>>(PF(P_G1_WE), PF(P_G1_AE), we1, EDGED, H1, FD);
  fold_aw<<<gb(EDGED), 256, 0, stream>>>(PF(P_G2_WE), PF(P_G2_AE), wee2, EDGED, 1, OUT);

  // ---- self-loop edge_attr (mean by dst) ----
  fill_f32<<<gb((long)NNODES * EDGED), 256, 0, stream>>>(sEA, 0.f, (long)NNODES * EDGED);
  fill_f32<<<gb(NNODES), 256, 0, stream>>>(cEA, 0.f, NNODES);
  edge_accum<<<gb((long)NEDGES * EDGED), 256, 0, stream>>>(ea, edst, sEA, cEA);
  build_ea_aug<<<gb((long)NAUG * EDGED), 256, 0, stream>>>(ea, sEA, cEA, eaug);

  // ---- GAT layer 1 ----
  gemm(stream, x, FD, PF(P_G1_W), HD1, 0, A1, HD1, NNODES, HD1, FD, nullptr, 0);
  gemm(stream, x, FD, ws1, H1, 0, asrc1, H1, NNODES, H1, FD, nullptr, 0);
  gemm(stream, x, FD, wd1, H1, 0, adst1, H1, NNODES, H1, FD, nullptr, 0);
  gemm(stream, eaug, EDGED, we1, H1, 0, ae1, H1, NAUG, H1, EDGED, nullptr, 0);
  fill_f32<<<gb((long)NNODES * H1), 256, 0, stream>>>(m1, -1e30f, (long)NNODES * H1);
  fill_f32<<<gb((long)NNODES * H1), 256, 0, stream>>>(den1, 0.f, (long)NNODES * H1);
  gat_alpha<<<gb((long)NAUG * H1), 256, 0, stream>>>(asrc1, adst1, ae1, esrc, edst,
                                                     NEDGES, NAUG, H1, alp1, m1);
  gat_expa<<<gb((long)NAUG * H1), 256, 0, stream>>>(alp1, m1, den1, edst, NEDGES, NAUG, H1);
  fill_f32<<<gb((long)NNODES * HD1), 256, 0, stream>>>(B1, 0.f, (long)NNODES * HD1);
  gat_scatter<<<gb((long)NAUG * HD1), 256, 0, stream>>>(A1, alp1, den1, esrc, edst,
                                                        NEDGES, NAUG, H1, FD, B1);
  bias_act<<<gb((long)NNODES * HD1), 256, 0, stream>>>(B1, PF(P_G1_B),
                                                       (long)NNODES * HD1, HD1, 2); // elu -> h1

  // ---- GAT layer 2 ----
  gemm(stream, B1, HD1, PF(P_G2_W), OUT, 0, h2pre, OUT, NNODES, OUT, HD1, nullptr, 0);
  gemm(stream, h2pre, OUT, PF(P_G2_AS), 1, 0, asrc2, 1, NNODES, 1, OUT, nullptr, 0);
  gemm(stream, h2pre, OUT, PF(P_G2_AD), 1, 0, adst2, 1, NNODES, 1, OUT, nullptr, 0);
  gemm(stream, eaug, EDGED, wee2, 1, 0, ae2, 1, NAUG, 1, EDGED, nullptr, 0);
  fill_f32<<<gb(NNODES), 256, 0, stream>>>(m2, -1e30f, NNODES);
  fill_f32<<<gb(NNODES), 256, 0, stream>>>(den2, 0.f, NNODES);
  gat_alpha<<<gb(NAUG), 256, 0, stream>>>(asrc2, adst2, ae2, esrc, edst,
                                          NEDGES, NAUG, 1, alp2, m2);
  gat_expa<<<gb(NAUG), 256, 0, stream>>>(alp2, m2, den2, edst, NEDGES, NAUG, 1);
  fill_f32<<<gb((long)NNODES * OUT), 256, 0, stream>>>(agg2, 0.f, (long)NNODES * OUT);
  gat_scatter<<<gb((long)NAUG * OUT), 256, 0, stream>>>(h2pre, alp2, den2, esrc, edst,
                                                        NEDGES, NAUG, 1, OUT, agg2);
  bias_act<<<gb((long)NNODES * OUT), 256, 0, stream>>>(agg2, PF(P_G2_B),
                                                       (long)NNODES * OUT, OUT, 2); // -> h2

  // ---- pool + mol linear ----
  pool_max<<<gb(BB * OUT), 256, 0, stream>>>(agg2, xg);
  gemm(stream, xg, OUT, PF(P_MOL_W), OUT, 0, xmol, OUT, BB, OUT, OUT, PF(P_MOL_B), 1);

  // ---- KG branches ----
  gather_rows<<<gb(BB * 128), 256, 0, stream>>>(PF(P_D_EMB), dkg, dgat, BB, 128);
  gather_rows<<<gb(BB * 128), 256, 0, stream>>>(PF(P_P_EMB), pkg, pgat, BB, 128);
  gemm(stream, dgat, 128, PF(P_D_W), OUT, 0, xdkg, OUT, BB, OUT, 128, PF(P_D_B), 1);
  gemm(stream, pgat, 128, PF(P_P_W), OUT, 0, xpkg, OUT, BB, OUT, 128, PF(P_P_B), 1);

  // ---- protein CNN (h2/agg2 and A1/B1 dead after pooling; reuse) ----
  float* y1 = agg2;  // 512*32*249 fits in 25600*256
  float* y3 = A1;    // 512*128*243
  float* y4 = B1;    // 512*128*241
  float* y5 = A1;    // 512*64*237 (y3 dead)
  float* y6 = B1;    // 512*32*231 (y4 dead)
  conv1_fused<<<BB, 256, 0, stream>>>(seq, PF(P_EMB_XT), PF(P_C1W), PF(P_C1B), y1);
  convw(stream, y1,  PF(P_C2W), PF(P_C2B), y2b, 32, 249, 64, 5, 245, 1);
  convw(stream, y2b, PF(P_C3W), PF(P_C3B), y3, 64, 245, 128, 3, 243, 1);
  convw(stream, y3,  PF(P_C4W), PF(P_C4B), y4, 128, 243, 128, 3, 241, 1);
  convw(stream, y4,  PF(P_C5W), PF(P_C5B), y5, 128, 241, 64, 5, 237, 1);
  convw(stream, y5,  PF(P_C6W), PF(P_C6B), y6, 64, 237, 32, 7, 231, 0);
  gemm(stream, y6, CNNDIM, PF(P_PROT_W), OUT, 0, xprot, OUT, BB, OUT, CNNDIM,
       PF(P_PROT_B), 1);

  // ---- fusion: 4-token MHA + residual LN ----
  build_xc<<<gb(BB * OUT), 256, 0, stream>>>(xmol, xdkg, xprot, xpkg, xc);
  gemm(stream, xc, OUT, PF(P_AIN_W), OUT, 1, qkv, 768, BB * 4, 768, OUT,
       PF(P_AIN_B), 0);                                     // x @ Wi.T + bi
  mha4<<<gb(BB * 16), 256, 0, stream>>>(qkv, attno);
  gemm(stream, attno, OUT, PF(P_AOUT_W), OUT, 1, proj, OUT, BB * 4, OUT, OUT,
       PF(P_AOUT_B), 0);                                    // o @ Wo.T + bo
  add_ln<<<gb(BB * 4), 256, 0, stream>>>(xc, proj, PF(P_N1_G), PF(P_N1_B), xcln);

  // ---- MLP head ----
  gemm(stream, xcln, 1024, PF(P_FC1_W), 1024, 0, z1, 1024, BB, 1024, 1024, PF(P_FC1_B), 1);
  gemm(stream, z1, 1024, PF(P_FC2_W), 512, 0, z2, 512, BB, 512, 1024, PF(P_FC2_B), 1);
  gemm(stream, z2, 512, PF(P_FC3_W), 256, 0, z3, 256, BB, 256, 512, PF(P_FC3_B), 1);
  gemm(stream, z3, 256, PF(P_OUT_W), 1, 0, outp, 1, BB, 1, 256, PF(P_OUT_B), 0);
}
